// AttentionBlock_21277267985129
// MI455X (gfx1250) — matrix operand order repaired and checked by reading, not compiled, not run on hardware
//
#include <hip/hip_runtime.h>

#define B_  4
#define C_  256
#define L_  4096
#define C8_ 32

typedef __attribute__((ext_vector_type(16))) __bf16 v16bf;
typedef __attribute__((ext_vector_type(8)))  float  v8f;

__device__ __forceinline__ unsigned short f2bf(float f) {
    unsigned int u = __float_as_uint(f);
    u += 0x7FFFu + ((u >> 16) & 1u);       // round-to-nearest-even
    return (unsigned short)(u >> 16);
}

// fast sigmoid: prefer hardware v_tanh_f32 (1 TRANS op) if the builtin exists,
// else v_exp_f32 + v_rcp_f32 (2 TRANS ops). Feeds a bf16 quantization, so the
// approximation error is negligible.
__device__ __forceinline__ float fast_sigmoid(float x) {
#if __has_builtin(__builtin_amdgcn_tanhf)
    return __builtin_fmaf(0.5f, __builtin_amdgcn_tanhf(0.5f * x), 0.5f);
#elif __has_builtin(__builtin_amdgcn_tanh_f32)
    return __builtin_fmaf(0.5f, __builtin_amdgcn_tanh_f32(0.5f * x), 0.5f);
#else
    float e = __expf(-x);
    return __builtin_amdgcn_rcpf(1.0f + e);
#endif
}

union BFrag {
    v16bf v;
    uint4 q[2];
};

struct KFrag { BFrag f[2]; };   // k fragments for one 32-wide j tile

// ---------------------------------------------------------------------------
// Kernel 1: q^T, k^T = (Wq x + bq)^T, (Wk x + bk)^T   stored bf16 [B][L][32]
// ---------------------------------------------------------------------------
__global__ __launch_bounds__(256) void qk_prep(
    const float* __restrict__ x,
    const float* __restrict__ Wq, const float* __restrict__ bq,
    const float* __restrict__ Wk, const float* __restrict__ bk,
    unsigned short* __restrict__ qt, unsigned short* __restrict__ kt)
{
    int b = blockIdx.y;
    int l = blockIdx.x * 256 + threadIdx.x;
    const float* xb = x + (size_t)b * C_ * L_ + l;

    float aq[32], ak[32];
#pragma unroll
    for (int d = 0; d < 32; ++d) { aq[d] = bq[d]; ak[d] = bk[d]; }

    for (int c0 = 0; c0 < C_; c0 += 32) {
        float xr[32];
#pragma unroll
        for (int cc = 0; cc < 32; ++cc) xr[cc] = xb[(size_t)(c0 + cc) * L_];
#pragma unroll
        for (int d = 0; d < 32; ++d) {
            const float* wqr = Wq + d * C_ + c0;   // uniform -> scalar loads
            const float* wkr = Wk + d * C_ + c0;
#pragma unroll
            for (int cc = 0; cc < 32; ++cc) {
                aq[d] = __builtin_fmaf(wqr[cc], xr[cc], aq[d]);
                ak[d] = __builtin_fmaf(wkr[cc], xr[cc], ak[d]);
            }
        }
    }
    unsigned int* qo = reinterpret_cast<unsigned int*>(qt + ((size_t)b * L_ + l) * 32);
    unsigned int* ko = reinterpret_cast<unsigned int*>(kt + ((size_t)b * L_ + l) * 32);
#pragma unroll
    for (int d2 = 0; d2 < 16; ++d2) {
        qo[d2] = (unsigned)f2bf(aq[2*d2]) | ((unsigned)f2bf(aq[2*d2+1]) << 16);
        ko[d2] = (unsigned)f2bf(ak[2*d2]) | ((unsigned)f2bf(ak[2*d2+1]) << 16);
    }
}

// ---------------------------------------------------------------------------
// Kernel 2: v = Wv x + bv   stored bf16 [B][C][L]
// ---------------------------------------------------------------------------
__global__ __launch_bounds__(256) void v_prep(
    const float* __restrict__ x,
    const float* __restrict__ Wv, const float* __restrict__ bv,
    unsigned short* __restrict__ vb)
{
    int b  = blockIdx.z;
    int o0 = blockIdx.y * 32;
    int l  = blockIdx.x * 256 + threadIdx.x;
    const float* xb = x + (size_t)b * C_ * L_ + l;

    float acc[32];
#pragma unroll
    for (int oo = 0; oo < 32; ++oo) acc[oo] = bv[o0 + oo];

    for (int c0 = 0; c0 < C_; c0 += 32) {
        float xr[32];
#pragma unroll
        for (int cc = 0; cc < 32; ++cc) xr[cc] = xb[(size_t)(c0 + cc) * L_];
#pragma unroll
        for (int oo = 0; oo < 32; ++oo) {
            const float* wr = Wv + (size_t)(o0 + oo) * C_ + c0;
#pragma unroll
            for (int cc = 0; cc < 32; ++cc)
                acc[oo] = __builtin_fmaf(wr[cc], xr[cc], acc[oo]);
        }
    }
#pragma unroll
    for (int oo = 0; oo < 32; ++oo)
        vb[((size_t)b * C_ + o0 + oo) * L_ + l] = f2bf(acc[oo]);
}

// ---------------------------------------------------------------------------
// k-fragment loader (one 32-wide j tile, two 16-col B fragments)
// ---------------------------------------------------------------------------
__device__ __forceinline__ KFrag load_k(
    const unsigned short* __restrict__ ktb, int jb, int hi, int m16)
{
    KFrag kf;
#pragma unroll
    for (int t = 0; t < 2; ++t) {
        const unsigned short* krow =
            ktb + (size_t)(jb + t * 16 + m16) * 32 + 16 * hi;
        kf.f[t].q[0] = *reinterpret_cast<const uint4*>(krow);
        kf.f[t].q[1] = *reinterpret_cast<const uint4*>(krow + 8);
    }
    return kf;
}

// ---------------------------------------------------------------------------
// S-tile producer: 16i x 32j tile of sigmoid(q^T k) -> bf16 in LDS.
// Truncating f32->bf16 (store of high 16 bits -> ds_store_b16_d16_hi).
// ---------------------------------------------------------------------------
__device__ __forceinline__ void s_tile(
    const KFrag& kf, const BFrag& aq, unsigned short* sbuf, int hi, int m16)
{
    v8f zero = {0.f,0.f,0.f,0.f,0.f,0.f,0.f,0.f};
#pragma unroll
    for (int t = 0; t < 2; ++t) {
        v8f s = __builtin_amdgcn_wmma_f32_16x16x32_bf16(
                    false, aq.v, false, kf.f[t].v, (short)0, zero, false, false);
#pragma unroll
        for (int r = 0; r < 8; ++r) {
            float sv = fast_sigmoid(s[r]);
            // D layout: row i = r + 8*hi, col j = t*16 + m16
            sbuf[(r + 8 * hi) * 32 + t * 16 + m16] =
                (unsigned short)(__float_as_uint(sv) >> 16);   // -> d16_hi store
        }
    }
}

// ---------------------------------------------------------------------------
// Kernel 3: fused  out = gamma * ( v @ sigmoid(q^T k)^T ) + x
// One wave per 16-wide i-tile; double-buffered S tile + 2-stage k prefetch:
// next tile's QK WMMA + sigmoid co-execute with current tile's 16 PV WMMAs,
// and k loads are issued a full iteration before their consuming WMMA.
// ---------------------------------------------------------------------------
__global__ __launch_bounds__(256) void attn_fused(
    const unsigned short* __restrict__ qt,
    const unsigned short* __restrict__ kt,
    const unsigned short* __restrict__ vb,
    const float* __restrict__ x,
    const float* __restrict__ gamma,
    float* __restrict__ out)
{
    __shared__ unsigned short slds[8][2][16 * 32];   // per-wave double buffer

    int wave = threadIdx.x >> 5;
    int lane = threadIdx.x & 31;
    int hi   = lane >> 4;          // 0: lanes 0-15, 1: lanes 16-31
    int m16  = lane & 15;
    int b    = blockIdx.y;
    int iBase = blockIdx.x * 128 + wave * 16;

    const unsigned short* qtb = qt + (size_t)b * L_ * 32;
    const unsigned short* ktb = kt + (size_t)b * L_ * 32;
    const unsigned short* vbb = vb + (size_t)b * C_ * L_;

    // A fragment (q^T tile, M=i, K=d=32): fixed across the whole j loop.
    BFrag aq;
    {
        const unsigned short* qrow = qtb + (size_t)(iBase + m16) * 32;
        aq.q[0] = *reinterpret_cast<const uint4*>(qrow + 16 * hi);
        aq.q[1] = *reinterpret_cast<const uint4*>(qrow + 16 * hi + 8);
    }

    v8f acc[16];
    v8f zero = {0.f,0.f,0.f,0.f,0.f,0.f,0.f,0.f};
#pragma unroll
    for (int ct = 0; ct < 16; ++ct) acc[ct] = zero;

    unsigned short* sb0 = slds[wave][0];
    unsigned short* sb1 = slds[wave][1];

    // prologue: S tile for jb = 0, then prefetch k for jb = 32
    KFrag kfA = load_k(ktb, 0, hi, m16);
    s_tile(kfA, aq, sb0, hi, m16);
    kfA = load_k(ktb, 32, hi, m16);
    asm volatile("" ::: "memory");

    auto body = [&](unsigned short* cur, unsigned short* nxt, int jb) {
        // B fragment for PV from previous tile: B^T[n=i][k=j] == S[i][j]
        BFrag bs;
        const unsigned short* srow = cur + m16 * 32 + 16 * hi;
        bs.q[0] = *reinterpret_cast<const uint4*>(srow);
        bs.q[1] = *reinterpret_cast<const uint4*>(srow + 8);

        // prefetch k fragments two tiles ahead
        KFrag kfB;
        if (jb + 64 < L_)
            kfB = load_k(ktb, jb + 64, hi, m16);

        // produce next S tile (QK WMMA + sigmoid) while PV WMMAs run below
        if (jb + 32 < L_)
            s_tile(kfA, aq, nxt, hi, m16);

        // ---- out(c,i) += v(c,j) * S^T(j,i), 16 c-tiles, K=32 ----
#pragma unroll
        for (int ct = 0; ct < 16; ++ct) {
            BFrag av;    // A fragment: M=c, K=j, from v[c][j] row-major
            const unsigned short* vrow =
                vbb + (size_t)(ct * 16 + m16) * L_ + jb + 16 * hi;
            av.q[0] = *reinterpret_cast<const uint4*>(vrow);
            av.q[1] = *reinterpret_cast<const uint4*>(vrow + 8);
            acc[ct] = __builtin_amdgcn_wmma_f32_16x16x32_bf16(
                          false, av.v, false, bs.v, (short)0, acc[ct], false, false);
        }
        kfA = kfB;
        // cross-iteration LDS store->load ordering (HW DS pipe is in-order)
        asm volatile("" ::: "memory");
    };

    for (int jb = 0; jb < L_; jb += 64) {
        body(sb0, sb1, jb);
        body(sb1, sb0, jb + 32);
    }

    // ---- epilogue: gamma * acc + x ----
    float g = gamma[0];
    const float* xb = x + (size_t)b * C_ * L_;
    float*       ob = out + (size_t)b * C_ * L_;
#pragma unroll
    for (int ct = 0; ct < 16; ++ct) {
#pragma unroll
        for (int r = 0; r < 8; ++r) {
            size_t idx = (size_t)(ct * 16 + r + 8 * hi) * L_ + iBase + m16;
            ob[idx] = g * acc[ct][r] + xb[idx];
        }
    }
}

// ---------------------------------------------------------------------------
extern "C" void kernel_launch(void* const* d_in, const int* in_sizes, int n_in,
                              void* d_out, int out_size, void* d_ws, size_t ws_size,
                              hipStream_t stream)
{
    const float* x     = (const float*)d_in[0];
    const float* Wq    = (const float*)d_in[1];
    const float* bq    = (const float*)d_in[2];
    const float* Wk    = (const float*)d_in[3];
    const float* bk    = (const float*)d_in[4];
    const float* Wv    = (const float*)d_in[5];
    const float* bv    = (const float*)d_in[6];
    const float* gamma = (const float*)d_in[7];
    float* out = (float*)d_out;

    // workspace: qt (1MB) | kt (1MB) | v bf16 (8MB)
    unsigned short* qt = (unsigned short*)d_ws;
    unsigned short* kt = qt + (size_t)B_ * L_ * 32;
    unsigned short* vb = kt + (size_t)B_ * L_ * 32;

    dim3 g1(L_ / 256, B_);
    qk_prep<<<g1, 256, 0, stream>>>(x, Wq, bq, Wk, bk, qt, kt);

    dim3 g2(L_ / 256, C_ / 32, B_);
    v_prep<<<g2, 256, 0, stream>>>(x, Wv, bv, vb);

    dim3 g3(L_ / 128, B_);   // 32 x 4 blocks, 8 waves/block, 1 i-tile per wave
    attn_fused<<<g3, 256, 0, stream>>>(qt, kt, vb, x, gamma, out);
}
